// TabNSA_73547019976847
// MI455X (gfx1250) — matrix operand order repaired and checked by reading, not compiled, not run on hardware
//
#include <hip/hip_runtime.h>
#include <hip/hip_bf16.h>

// ---- problem constants (from reference) ----
#define B_    256
#define N_    192
#define DIM_  64
#define DH_   24
#define BLK_  16
#define NBLK_ 12
#define NSEL_ 4
#define DFF_  256
#define HID_  32
#define OUT_  2
#define SCALE_ 0.2041241452319315f   // 24^-0.5
#define NEG_  -1e30f

typedef __attribute__((ext_vector_type(16))) _Float16 v16h;
typedef __attribute__((ext_vector_type(8)))  _Float16 v8h;
typedef __attribute__((ext_vector_type(8)))  float    v8f;

__device__ __forceinline__ float gelu_f(float x) {
    return 0.5f * x * (1.0f + erff(x * 0.70710678118654752f));
}

// Build a 16x32 f16 A fragment from two contiguous 8-half chunks.
// Element e<8  -> chunk 0 [e], element e>=8 -> chunk 1 [e-8]
// (callers pass base + half*16 and base + half*16 + 8: K = half*16 + e)
__device__ __forceinline__ v16h load_a_frag(const _Float16* c0, const _Float16* c1) {
    v16h a;
    *(v8h*)&a       = *(const v8h*)c0;
    *((v8h*)&a + 1) = *(const v8h*)c1;
    return a;
}

// =====================================================================
// Weight transpose + f32->f16 convert:  WT[c*K + k] = (f16) W[k*Ncol + c]
// =====================================================================
__global__ __launch_bounds__(256) void convert_w_kernel(
    const float* __restrict__ W, _Float16* __restrict__ WT, int K, int Ncol)
{
    int i = blockIdx.x * 256 + threadIdx.x;
    if (i < K * Ncol) {
        int k = i / Ncol, c = i % Ncol;
        WT[c * K + k] = (_Float16)W[i];
    }
}

// =====================================================================
// Kernel 1: embed + rmsnorm + qkv + gates.  one row (b,n) per 64-thread block
// =====================================================================
__global__ __launch_bounds__(64) void embed_qkv_kernel(
    const float* __restrict__ x, const float* __restrict__ We,
    const float* __restrict__ be, const float* __restrict__ rms_w,
    const float* __restrict__ Wqkv, const float* __restrict__ Wg,
    const float* __restrict__ bg,
    float* __restrict__ e_out, float* __restrict__ q, float* __restrict__ k,
    float* __restrict__ v, float* __restrict__ g3)
{
    int row = blockIdx.x;           // b*N + n
    int tid = threadIdx.x;          // 0..63 == dim
    __shared__ float s_h[DIM_];
    __shared__ float s_red[DIM_];

    float xv = x[row];
    float ev = xv * We[tid] + be[tid];
    s_red[tid] = ev * ev;
    e_out[(size_t)row * DIM_ + tid] = ev;
    __syncthreads();
    for (int off = 32; off > 0; off >>= 1) {
        if (tid < off) s_red[tid] += s_red[tid + off];
        __syncthreads();
    }
    float rms = rsqrtf(s_red[0] / (float)DIM_ + 1e-6f);
    s_h[tid] = ev * rms * rms_w[tid];
    __syncthreads();

    for (int o = tid; o < 3 * DH_; o += 64) {
        float acc = 0.f;
        for (int kk = 0; kk < DIM_; ++kk) acc += s_h[kk] * Wqkv[kk * (3 * DH_) + o];
        int which = o / DH_, j = o % DH_;
        float* dst = (which == 0) ? q : (which == 1) ? k : v;
        dst[(size_t)row * DH_ + j] = acc;
    }
    if (tid < 3) {
        float acc = bg[tid];
        for (int kk = 0; kk < DIM_; ++kk) acc += s_h[kk] * Wg[kk * 3 + tid];
        g3[(size_t)row * 3 + tid] = 1.0f / (1.0f + expf(-acc));
    }
}

// =====================================================================
// Kernel 2: block compression  (one (b,blk) per 64-thread block)
// =====================================================================
__global__ __launch_bounds__(64) void compress_kernel(
    const float* __restrict__ k, const float* __restrict__ v,
    const float* __restrict__ k_pos, const float* __restrict__ v_pos,
    const float* __restrict__ Wk, const float* __restrict__ bk,
    const float* __restrict__ Wv, const float* __restrict__ bv,
    float* __restrict__ ck, float* __restrict__ cv)
{
    int bb  = blockIdx.x;            // b*NBLK + blk
    int b   = bb / NBLK_, blk = bb % NBLK_;
    int tid = threadIdx.x;
    __shared__ float s_kb[BLK_ * DH_];
    __shared__ float s_vb[BLK_ * DH_];
    for (int i = tid; i < BLK_ * DH_; i += 64) {
        int t = i / DH_, d = i % DH_;
        size_t ridx = ((size_t)b * N_ + blk * BLK_ + t) * DH_ + d;
        s_kb[i] = k[ridx] + k_pos[i];
        s_vb[i] = v[ridx] + v_pos[i];
    }
    __syncthreads();
    if (tid < DH_) {
        float ak = bk[tid], av = bv[tid];
        for (int kk = 0; kk < BLK_ * DH_; ++kk) {
            ak += s_kb[kk] * Wk[kk * DH_ + tid];
            av += s_vb[kk] * Wv[kk * DH_ + tid];
        }
        ck[(size_t)bb * DH_ + tid] = ak;
        cv[(size_t)bb * DH_ + tid] = av;
    }
}

// =====================================================================
// Kernel 3: compressed attention + top-4 selection.  one row per wave
// =====================================================================
__global__ __launch_bounds__(32) void cattn_kernel(
    const float* __restrict__ q, const float* __restrict__ ck,
    const float* __restrict__ cv, const float* __restrict__ mem_kv,
    float* __restrict__ cout, int* __restrict__ sidx, int* __restrict__ smask)
{
    int row = blockIdx.x;
    int b = row / N_, n = row % N_;
    int tid = threadIdx.x;
    __shared__ float s_q[DH_];
    __shared__ float s_a[NBLK_ + 1];
    if (tid < DH_) s_q[tid] = q[(size_t)row * DH_ + tid];
    __syncthreads();
    if (tid < NBLK_ + 1) {
        const float* kk = (tid == 0) ? mem_kv
                                     : (ck + ((size_t)b * NBLK_ + tid - 1) * DH_);
        float s = 0.f;
        for (int d = 0; d < DH_; ++d) s += s_q[d] * kk[d];
        s *= SCALE_;
        int seq = (tid == 0) ? -1 : tid * BLK_ - 1;
        s_a[tid] = (n >= seq) ? s : NEG_;
    }
    __syncthreads();
    if (tid == 0) {
        float mx = -1e38f;
        for (int j = 0; j <= NBLK_; ++j) mx = fmaxf(mx, s_a[j]);
        float sm = 0.f;
        for (int j = 0; j <= NBLK_; ++j) { float e2 = expf(s_a[j] - mx); s_a[j] = e2; sm += e2; }
        float inv = 1.0f / sm;
        for (int j = 0; j <= NBLK_; ++j) s_a[j] *= inv;
    }
    __syncthreads();
    if (tid < DH_) {
        float acc = 0.f;
        for (int j = 0; j <= NBLK_; ++j) {
            const float* vv = (j == 0) ? (mem_kv + DH_)
                                       : (cv + ((size_t)b * NBLK_ + j - 1) * DH_);
            acc += s_a[j] * vv[tid];
        }
        cout[(size_t)row * DH_ + tid] = acc;
    }
    if (tid == 0) {
        int used[NBLK_]; for (int j = 0; j < NBLK_; ++j) used[j] = 0;
        for (int s = 0; s < NSEL_; ++s) {
            float best = -1e38f; int bi = 0;
            for (int j = 0; j < NBLK_; ++j)
                if (!used[j] && s_a[j + 1] > best) { best = s_a[j + 1]; bi = j; }
            used[bi] = 1;
            sidx [(size_t)row * 5 + s] = bi;
            smask[(size_t)row * 5 + s] = (best > 1e-10f) ? 1 : 0;
        }
        sidx [(size_t)row * 5 + 4] = n / BLK_;
        smask[(size_t)row * 5 + 4] = 1;
    }
}

// =====================================================================
// Kernel 4: RoPE for q and k.  one row per wave
// =====================================================================
__global__ __launch_bounds__(32) void rope_kernel(
    const float* __restrict__ q, const float* __restrict__ k,
    float* __restrict__ qr, float* __restrict__ kr)
{
    int row = blockIdx.x; int n = row % N_;
    int tid = threadIdx.x;
    if (tid < DH_ / 2) {
        float inv = powf(10000.0f, -(2.0f * tid) / (float)DH_);
        float ang = (float)n * inv;
        float c = cosf(ang), s = sinf(ang);
        size_t base = (size_t)row * DH_ + tid * 2;
        float q0 = q[base], q1 = q[base + 1];
        qr[base]     = q0 * c - q1 * s;
        qr[base + 1] = q1 * c + q0 * s;
        float k0 = k[base], k1 = k[base + 1];
        kr[base]     = k0 * c - k1 * s;
        kr[base + 1] = k1 * c + k0 * s;
    }
}

// =====================================================================
// Kernel 5: fine attention + gate combine + x1 = ao @ W_comb
// (WIN=1 => sliding-window output degenerates to v itself)
// =====================================================================
__global__ __launch_bounds__(128) void fattn_kernel(
    const float* __restrict__ qr, const float* __restrict__ kr,
    const float* __restrict__ v, const float* __restrict__ cout,
    const float* __restrict__ g3, const int* __restrict__ sidx,
    const int* __restrict__ smask, const float* __restrict__ W_comb,
    float* __restrict__ x1)
{
    int row = blockIdx.x;
    int b = row / N_, n = row % N_;
    int tid = threadIdx.x;
    __shared__ float s_q[DH_];
    __shared__ float s_sim[5 * BLK_];
    __shared__ float s_ao[DH_];
    __shared__ int   s_idx[5], s_msk[5];
    if (tid < DH_) s_q[tid] = qr[(size_t)row * DH_ + tid];
    if (tid < 5) { s_idx[tid] = sidx[(size_t)row * 5 + tid];
                   s_msk[tid] = smask[(size_t)row * 5 + tid]; }
    __syncthreads();
    if (tid < 5 * BLK_) {
        int sb = tid / BLK_, t = tid % BLK_;
        int pos = s_idx[sb] * BLK_ + t;
        float sim = NEG_;
        if (s_msk[sb] && pos <= n) {
            const float* kk = kr + ((size_t)b * N_ + pos) * DH_;
            float s = 0.f;
            for (int d = 0; d < DH_; ++d) s += s_q[d] * kk[d];
            sim = s * SCALE_;
        }
        s_sim[tid] = sim;
    }
    __syncthreads();
    if (tid == 0) {
        float mx = -1e38f;
        for (int j = 0; j < 5 * BLK_; ++j) mx = fmaxf(mx, s_sim[j]);
        float sm = 0.f;
        for (int j = 0; j < 5 * BLK_; ++j) { float e2 = expf(s_sim[j] - mx); s_sim[j] = e2; sm += e2; }
        float inv = 1.0f / sm;
        for (int j = 0; j < 5 * BLK_; ++j) s_sim[j] *= inv;
    }
    __syncthreads();
    if (tid < DH_) {
        float acc = 0.f;
        for (int j = 0; j < 5 * BLK_; ++j) {
            int pos = s_idx[j / BLK_] * BLK_ + (j % BLK_);
            acc += s_sim[j] * v[((size_t)b * N_ + pos) * DH_ + tid];
        }
        float g0 = g3[(size_t)row * 3 + 0];
        float g1 = g3[(size_t)row * 3 + 1];
        float g2 = g3[(size_t)row * 3 + 2];
        float sout = v[(size_t)row * DH_ + tid];      // WIN=1 => identity attention
        s_ao[tid] = g0 * cout[(size_t)row * DH_ + tid] + g1 * acc + g2 * sout;
    }
    __syncthreads();
    if (tid < DIM_) {
        float acc = 0.f;
        for (int kk = 0; kk < DH_; ++kk) acc += s_ao[kk] * W_comb[kk * DIM_ + tid];
        x1[(size_t)row * DIM_ + tid] = acc;
    }
}

// =====================================================================
// Kernel 6: token mixer (WMMA).  grid = B*4; each WG: 16 dims of one batch.
// x2 = e + ( gelu(ln1(e)^T @ Wt1 + bt1) @ Wt2 + bt2 )^T
// Weights pre-transposed to f16:  WT[c*K + k]
// =====================================================================
__global__ __launch_bounds__(256) void token_mixer_kernel(
    const float* __restrict__ e, const float* __restrict__ ln1_g,
    const float* __restrict__ ln1_b,
    const _Float16* __restrict__ Wt1T, const float* __restrict__ bt1,
    const _Float16* __restrict__ Wt2T, const float* __restrict__ bt2,
    float* __restrict__ x2)
{
    int wg = blockIdx.x;
    int b  = wg >> 2;
    int d0 = (wg & 3) * 16;
    int tid = threadIdx.x, lane = tid & 31, wave = tid >> 5;
    int m = lane & 15, half = lane >> 4;

    __shared__ float    s_mean[N_], s_rstd[N_];
    __shared__ _Float16 s_a[16 * N_];        // LN'd A tile (f16): [dim][token]
    __shared__ _Float16 s_u[16 * DFF_];      // gelu(stage1) tile (f16)

    // LN stats per token
    for (int n = tid; n < N_; n += 256) {
        const float* row = e + ((size_t)b * N_ + n) * DIM_;
        float s = 0.f, ss = 0.f;
        for (int d = 0; d < DIM_; ++d) { float vv = row[d]; s += vv; ss += vv * vv; }
        float mn = s / (float)DIM_;
        s_mean[n] = mn;
        s_rstd[n] = rsqrtf(ss / (float)DIM_ - mn * mn + 1e-5f);
    }
    __syncthreads();

    // Stage A tile into LDS (coalesced reads of e: consecutive tid -> consecutive dim)
    for (int i = tid; i < 16 * N_; i += 256) {
        int mloc = i & 15;
        int ka   = i >> 4;
        float av = (e[((size_t)b * N_ + ka) * DIM_ + d0 + mloc] - s_mean[ka])
                   * s_rstd[ka] * ln1_g[d0 + mloc] + ln1_b[d0 + mloc];
        s_a[mloc * N_ + ka] = (_Float16)av;
    }
    __syncthreads();

    // hoist stage-1 A fragments (identical across output tiles)
    v16h afr1[N_ / 32];
#pragma unroll
    for (int s = 0; s < N_ / 32; ++s) {
        const _Float16* base = s_a + m * N_ + s * 32 + half * 16;
        afr1[s] = load_a_frag(base, base + 8);
    }

    // stage 1: (16 x 192) @ (192 x 256) -> gelu -> LDS f16
    for (int t = wave; t < DFF_ / 16; t += 8) {
        int n0 = t * 16;
        v8f c = {};
#pragma unroll
        for (int s = 0; s < N_ / 32; ++s) {
            v16h bm = *(const v16h*)(Wt1T + (size_t)(n0 + m) * N_ + s * 32 + half * 16);
            c = __builtin_amdgcn_wmma_f32_16x16x32_f16(false, afr1[s], false, bm,
                                                       (short)0, c, false, false);
        }
#pragma unroll
        for (int r = 0; r < 8; ++r) {
            int mm = r + half * 8;
            float vv = c[r] + bt1[n0 + m];
            s_u[mm * DFF_ + n0 + m] = (_Float16)gelu_f(vv);
        }
    }
    __syncthreads();

    // hoist stage-2 A fragments
    v16h afr2[DFF_ / 32];
#pragma unroll
    for (int s = 0; s < DFF_ / 32; ++s) {
        const _Float16* base = s_u + m * DFF_ + s * 32 + half * 16;
        afr2[s] = load_a_frag(base, base + 8);
    }

    // stage 2: (16 x 256) @ (256 x 192) + bt2 ; x2 = e + result^T
    for (int t = wave; t < N_ / 16; t += 8) {
        int n0 = t * 16;
        v8f c = {};
#pragma unroll
        for (int s = 0; s < DFF_ / 32; ++s) {
            v16h bm = *(const v16h*)(Wt2T + (size_t)(n0 + m) * DFF_ + s * 32 + half * 16);
            c = __builtin_amdgcn_wmma_f32_16x16x32_f16(false, afr2[s], false, bm,
                                                       (short)0, c, false, false);
        }
#pragma unroll
        for (int r = 0; r < 8; ++r) {
            int mm = r + half * 8;       // output dim within tile
            int nn = n0 + m;             // token index
            size_t idx = ((size_t)b * N_ + nn) * DIM_ + d0 + mm;
            x2[idx] = e[idx] + c[r] + bt2[nn];
        }
    }
}

// =====================================================================
// Kernel 7: feature MLP (WMMA).  grid = B*N/16; 16 rows per WG.
// x2 += gelu(ln2(x2) @ Wf1 + bf1) @ Wf2 + bf2
// =====================================================================
__global__ __launch_bounds__(256) void feat_mlp_kernel(
    const float* __restrict__ ln2_g, const float* __restrict__ ln2_b,
    const _Float16* __restrict__ Wf1T, const float* __restrict__ bf1,
    const _Float16* __restrict__ Wf2T, const float* __restrict__ bf2,
    float* __restrict__ x2)
{
    size_t r0 = (size_t)blockIdx.x * 16;
    int tid = threadIdx.x, lane = tid & 31, wave = tid >> 5;
    int m = lane & 15, half = lane >> 4;

    __shared__ float    s_x[16 * DIM_];
    __shared__ _Float16 s_f[16 * DIM_];
    __shared__ _Float16 s_u[16 * DFF_];

    for (int i = tid; i < 16 * DIM_; i += 256) s_x[i] = x2[r0 * DIM_ + i];
    __syncthreads();
    if (tid < 16) {
        float s = 0.f, ss = 0.f;
        for (int d = 0; d < DIM_; ++d) { float vv = s_x[tid * DIM_ + d]; s += vv; ss += vv * vv; }
        float mn = s / (float)DIM_;
        float rstd = rsqrtf(ss / (float)DIM_ - mn * mn + 1e-5f);
        for (int d = 0; d < DIM_; ++d)
            s_f[tid * DIM_ + d] =
                (_Float16)((s_x[tid * DIM_ + d] - mn) * rstd * ln2_g[d] + ln2_b[d]);
    }
    __syncthreads();

    // hoist stage-1 A fragments
    v16h afr1[DIM_ / 32];
#pragma unroll
    for (int s = 0; s < DIM_ / 32; ++s) {
        const _Float16* base = s_f + m * DIM_ + s * 32 + half * 16;
        afr1[s] = load_a_frag(base, base + 8);
    }

    // stage 1: (16 x 64) @ (64 x 256) -> gelu -> LDS
    for (int t = wave; t < DFF_ / 16; t += 8) {
        int n0 = t * 16;
        v8f c = {};
#pragma unroll
        for (int s = 0; s < DIM_ / 32; ++s) {
            v16h bm = *(const v16h*)(Wf1T + (size_t)(n0 + m) * DIM_ + s * 32 + half * 16);
            c = __builtin_amdgcn_wmma_f32_16x16x32_f16(false, afr1[s], false, bm,
                                                       (short)0, c, false, false);
        }
#pragma unroll
        for (int r = 0; r < 8; ++r) {
            int mm = r + half * 8;
            float vv = c[r] + bf1[n0 + m];
            s_u[mm * DFF_ + n0 + m] = (_Float16)gelu_f(vv);
        }
    }
    __syncthreads();

    // stage 2: (16 x 256) @ (256 x 64) + bf2 ; residual add, write back
    if (wave < DIM_ / 16) {
        v16h afr2[DFF_ / 32];
#pragma unroll
        for (int s = 0; s < DFF_ / 32; ++s) {
            const _Float16* base = s_u + m * DFF_ + s * 32 + half * 16;
            afr2[s] = load_a_frag(base, base + 8);
        }
        int n0 = wave * 16;
        v8f c = {};
#pragma unroll
        for (int s = 0; s < DFF_ / 32; ++s) {
            v16h bm = *(const v16h*)(Wf2T + (size_t)(n0 + m) * DFF_ + s * 32 + half * 16);
            c = __builtin_amdgcn_wmma_f32_16x16x32_f16(false, afr2[s], false, bm,
                                                       (short)0, c, false, false);
        }
#pragma unroll
        for (int r = 0; r < 8; ++r) {
            int mm = r + half * 8;
            x2[(r0 + mm) * DIM_ + n0 + m] = s_x[mm * DIM_ + n0 + m] + c[r] + bf2[n0 + m];
        }
    }
}

// =====================================================================
// Kernel 8: mean over tokens + 2-layer head.  one batch per 64-thread block
// =====================================================================
__global__ __launch_bounds__(64) void head_kernel(
    const float* __restrict__ x1, const float* __restrict__ x2,
    const float* __restrict__ Wh1, const float* __restrict__ bh1,
    const float* __restrict__ Wh2, const float* __restrict__ bh2,
    float* __restrict__ out)
{
    int b = blockIdx.x, tid = threadIdx.x;
    __shared__ float s_s[DIM_], s_h[HID_];
    float acc = 0.f;
    for (int n = 0; n < N_; ++n) {
        size_t idx = ((size_t)b * N_ + n) * DIM_ + tid;
        acc += x1[idx] + x2[idx];
    }
    s_s[tid] = acc / (float)N_;
    __syncthreads();
    if (tid < HID_) {
        float a = bh1[tid];
        for (int kk = 0; kk < DIM_; ++kk) a += s_s[kk] * Wh1[kk * HID_ + tid];
        s_h[tid] = gelu_f(a);
    }
    __syncthreads();
    if (tid < OUT_) {
        float a = bh2[tid];
        for (int kk = 0; kk < HID_; ++kk) a += s_h[kk] * Wh2[kk * OUT_ + tid];
        out[(size_t)b * OUT_ + tid] = a;
    }
}

// =====================================================================
extern "C" void kernel_launch(void* const* d_in, const int* in_sizes, int n_in,
                              void* d_out, int out_size, void* d_ws, size_t ws_size,
                              hipStream_t stream)
{
    (void)in_sizes; (void)n_in; (void)out_size; (void)ws_size;
    const float* x       = (const float*)d_in[0];
    const float* W_embed = (const float*)d_in[1];
    const float* b_embed = (const float*)d_in[2];
    const float* rms_w   = (const float*)d_in[3];
    const float* W_qkv   = (const float*)d_in[4];
    const float* k_pos   = (const float*)d_in[5];
    const float* v_pos   = (const float*)d_in[6];
    const float* W_kcmp  = (const float*)d_in[7];
    const float* b_kcmp  = (const float*)d_in[8];
    const float* W_vcmp  = (const float*)d_in[9];
    const float* b_vcmp  = (const float*)d_in[10];
    const float* mem_kv  = (const float*)d_in[11];
    const float* W_gates = (const float*)d_in[12];
    const float* b_gates = (const float*)d_in[13];
    const float* W_comb  = (const float*)d_in[14];
    const float* ln1_g   = (const float*)d_in[15];
    const float* ln1_b   = (const float*)d_in[16];
    const float* Wt1     = (const float*)d_in[17];
    const float* bt1     = (const float*)d_in[18];
    const float* Wt2     = (const float*)d_in[19];
    const float* bt2     = (const float*)d_in[20];
    const float* ln2_g   = (const float*)d_in[21];
    const float* ln2_b   = (const float*)d_in[22];
    const float* Wf1     = (const float*)d_in[23];
    const float* bf1     = (const float*)d_in[24];
    const float* Wf2     = (const float*)d_in[25];
    const float* bf2     = (const float*)d_in[26];
    const float* Wh1     = (const float*)d_in[27];
    const float* bh1     = (const float*)d_in[28];
    const float* Wh2     = (const float*)d_in[29];
    const float* bh2     = (const float*)d_in[30];
    float* out = (float*)d_out;

    // ---- workspace layout (floats) ----
    float* ws = (float*)d_ws;
    size_t off = 0;
    float* e    = ws + off; off += (size_t)B_ * N_ * DIM_;
    float* q    = ws + off; off += (size_t)B_ * N_ * DH_;
    float* k    = ws + off; off += (size_t)B_ * N_ * DH_;
    float* v    = ws + off; off += (size_t)B_ * N_ * DH_;
    float* qr   = ws + off; off += (size_t)B_ * N_ * DH_;
    float* kr   = ws + off; off += (size_t)B_ * N_ * DH_;
    float* cout = ws + off; off += (size_t)B_ * N_ * DH_;
    float* g3   = ws + off; off += (size_t)B_ * N_ * 3;
    float* ck   = ws + off; off += (size_t)B_ * NBLK_ * DH_;
    float* cv   = ws + off; off += (size_t)B_ * NBLK_ * DH_;
    float* x1   = ws + off; off += (size_t)B_ * N_ * DIM_;
    float* x2   = ws + off; off += (size_t)B_ * N_ * DIM_;
    int* sidx   = (int*)(ws + off); off += (size_t)B_ * N_ * 5;
    int* smask  = (int*)(ws + off); off += (size_t)B_ * N_ * 5;
    off = (off + 15) & ~(size_t)15;            // 64B align for f16 vector loads
    _Float16* wt1t = (_Float16*)(ws + off); off += (size_t)N_ * DFF_ / 2;   // 192x256 halves
    _Float16* wt2t = (_Float16*)(ws + off); off += (size_t)DFF_ * N_ / 2;
    _Float16* wf1t = (_Float16*)(ws + off); off += (size_t)DIM_ * DFF_ / 2;
    _Float16* wf2t = (_Float16*)(ws + off); off += (size_t)DFF_ * DIM_ / 2;

    const int ROWS = B_ * N_;

    // one-time (per launch) weight transpose+convert to f16
    convert_w_kernel<<<(N_ * DFF_ + 255) / 256, 256, 0, stream>>>(Wt1, wt1t, N_, DFF_);
    convert_w_kernel<<<(DFF_ * N_ + 255) / 256, 256, 0, stream>>>(Wt2, wt2t, DFF_, N_);
    convert_w_kernel<<<(DIM_ * DFF_ + 255) / 256, 256, 0, stream>>>(Wf1, wf1t, DIM_, DFF_);
    convert_w_kernel<<<(DFF_ * DIM_ + 255) / 256, 256, 0, stream>>>(Wf2, wf2t, DFF_, DIM_);

    embed_qkv_kernel<<<ROWS, 64, 0, stream>>>(x, W_embed, b_embed, rms_w,
                                              W_qkv, W_gates, b_gates,
                                              e, q, k, v, g3);
    compress_kernel<<<B_ * NBLK_, 64, 0, stream>>>(k, v, k_pos, v_pos,
                                                   W_kcmp, b_kcmp, W_vcmp, b_vcmp,
                                                   ck, cv);
    cattn_kernel<<<ROWS, 32, 0, stream>>>(q, ck, cv, mem_kv, cout, sidx, smask);
    rope_kernel<<<ROWS, 32, 0, stream>>>(q, k, qr, kr);
    fattn_kernel<<<ROWS, 128, 0, stream>>>(qr, kr, v, cout, g3, sidx, smask,
                                           W_comb, x1);
    token_mixer_kernel<<<B_ * 4, 256, 0, stream>>>(e, ln1_g, ln1_b,
                                                   wt1t, bt1, wt2t, bt2, x2);
    feat_mlp_kernel<<<ROWS / 16, 256, 0, stream>>>(ln2_g, ln2_b,
                                                   wf1t, bf1, wf2t, bf2, x2);
    head_kernel<<<B_, 64, 0, stream>>>(x1, x2, Wh1, bh1, Wh2, bh2, out);
}
